// UBlock_6030134084118
// MI455X (gfx1250) — matrix operand order repaired and checked by reading, not compiled, not run on hardware
//
#include <hip/hip_runtime.h>
#include <hip/hip_bf16.h>

#define EPSV 1e-4f

typedef __attribute__((ext_vector_type(16))) __bf16    v16bf;
typedef __attribute__((ext_vector_type(8)))  float     v8f;
typedef __attribute__((ext_vector_type(8)))  unsigned  v8u;

// ---------- bf16 split helpers (fp32 ~= hi + lo emulation) ----------
__device__ __forceinline__ float bf_trunc_f(float f) {
    unsigned u = __builtin_bit_cast(unsigned, f) & 0xFFFF0000u;
    return __builtin_bit_cast(float, u);
}
__device__ __forceinline__ unsigned short f_to_bfu(float f) {
    return (unsigned short)(__builtin_bit_cast(unsigned, f) >> 16);
}

// =====================================================================
// Weight repack: (K, Cin, Cout) f32 -> per-lane WMMA B-fragments, bf16
// hi/lo split.  Layout (ushort index):
//   (((koff*nkc + kc)*nnt + t)*32 + lane)*32  + [0..15]=Bh, [16..31]=Bl
// where fragment element i holds K = kc*32 + 8*(lane/16) + (i<8 ? i : i+8),
// column N = t*16 + lane%16  (matches ISA 16-bit B layout).
// =====================================================================
__global__ void repack_w_kernel(const float* __restrict__ W, unsigned short* __restrict__ out,
                                int K, int Cin, int Cout)
{
    long long id = (long long)blockIdx.x * blockDim.x + threadIdx.x;
    long long total = (long long)K * Cin * Cout;
    if (id >= total) return;
    int i    = (int)(id & 15);
    long long r = id >> 4;
    int lane = (int)(r & 31); r >>= 5;
    int nnt = Cout >> 4;
    int t  = (int)(r % nnt); r /= nnt;
    int nkc = Cin >> 5;
    int kc = (int)(r % nkc);
    int koff = (int)(r / nkc);
    int half = lane >> 4, ln = lane & 15;
    int kk  = kc * 32 + half * 8 + ((i < 8) ? i : i + 8);
    int col = t * 16 + ln;
    float w  = W[((long long)koff * Cin + kk) * Cout + col];
    float wh = bf_trunc_f(w);
    long long base = (id >> 4) * 32;
    out[base + i]      = f_to_bfu(wh);
    out[base + 16 + i] = f_to_bfu(w - wh);
}

// =====================================================================
// Sparse gather-GEMM-scatter conv.  One wave32 per 16(M)x16(N) tile per
// kernel offset; K accumulated in chunks of 32 with 3x
// V_WMMA_F32_16X16X32_BF16 (hi*hi + hi*lo + lo*hi ~= fp32).
// Features arrive as packed bf16 hi/lo planes (from bn_relu); weights
// arrive pre-packed in fragment order (repack_w_kernel).
// A-frag: lane L -> row M=L%16, dword r -> K pair:
//   r<4: K=2r+8*half ; r>=4: K=2r+8+8*half  (pair (K,K+1) = one dword)
// D: VGPR j -> row M = j + 8*half, col N = L%16
// =====================================================================
__global__ __launch_bounds__(128)
void sparse_conv_wmma(const unsigned short* __restrict__ xh, size_t plane,
                      const unsigned short* __restrict__ wfrag,
                      const int* __restrict__ pin, const int* __restrict__ pout,
                      float* __restrict__ out, int P, int Cin, int Cout,
                      const int* __restrict__ ninp, const int* __restrict__ noutp)
{
    const int lane = threadIdx.x;        // 0..31
    const int half = lane >> 4;
    const int koff = blockIdx.y;
    const int m0   = (blockIdx.x * blockDim.y + threadIdx.y) * 16;
    const int tnt  = blockIdx.z;
    const int nb   = tnt * 16;
    const int n_in  = ninp[0];
    const int n_out = noutp[0];
    const int nkc = Cin >> 5;
    const int nnt = Cout >> 4;

    const size_t pbase = (size_t)koff * P;
    int m  = m0 + (lane & 15);
    int mc = m < P ? m : P - 1;
    int gidx = pin[pbase + mc];
    bool gv = (m < P) && ((unsigned)gidx < (unsigned)n_in);   // pad rows == n_in
    const unsigned short* xrh = xh + (size_t)gidx * Cin;
    const unsigned short* xrl = xrh + plane;

    v8f acc = {};
    for (int kc = 0; kc < nkc; ++kc) {
        const int kb = kc * 32;
        v8u au = {}, alu = {};
        if (gv) {
#pragma unroll
            for (int r = 0; r < 8; ++r) {
                int kk = kb + ((r < 4) ? (2 * r) : (2 * r + 8)) + half * 8;
                au[r]  = *(const unsigned*)(xrh + kk);
                alu[r] = *(const unsigned*)(xrl + kk);
            }
        }
        v16bf Ah = __builtin_bit_cast(v16bf, au);
        v16bf Al = __builtin_bit_cast(v16bf, alu);
        size_t fb = ((((size_t)koff * nkc + kc) * nnt + tnt) * 32 + lane) * 32;
        v16bf Bh = *(const v16bf*)(wfrag + fb);
        v16bf Bl = *(const v16bf*)(wfrag + fb + 16);
        acc = __builtin_amdgcn_wmma_f32_16x16x32_bf16(false, Ah, false, Bh, (short)0, acc, false, false);
        acc = __builtin_amdgcn_wmma_f32_16x16x32_bf16(false, Ah, false, Bl, (short)0, acc, false, false);
        acc = __builtin_amdgcn_wmma_f32_16x16x32_bf16(false, Al, false, Bh, (short)0, acc, false, false);
    }
#pragma unroll
    for (int j = 0; j < 8; ++j) {
        int mrow = m0 + j + 8 * half;
        if (mrow < P) {
            int o = pout[pbase + mrow];
            if ((unsigned)o < (unsigned)n_out)                // pad rows == n_out
                atomicAdd(out + (size_t)o * Cout + nb + (lane & 15), acc[j]);
        }
    }
}

// Dense GEMM (cat @ Wsc) from f32 input, in-kernel bf16 split, plain store.
__global__ __launch_bounds__(128)
void dense_gemm_wmma(const float* __restrict__ x, const float* __restrict__ Wm,
                     float* __restrict__ out, int M, int Cin, int Cout)
{
    const int lane = threadIdx.x;
    const int half = lane >> 4;
    const int ln   = lane & 15;
    const int m0   = (blockIdx.x * blockDim.y + threadIdx.y) * 16;
    const int nb   = blockIdx.z * 16;

    int m = m0 + ln;
    bool gv = (m < M);
    const float* xr = x + (size_t)(gv ? m : 0) * Cin;

    v8f acc = {};
    for (int kb = 0; kb < Cin; kb += 32) {
        v16bf Ah, Al, Bh, Bl;
#pragma unroll
        for (int r = 0; r < 8; ++r) {
            int kk = kb + ((r < 4) ? (2 * r) : (2 * r + 8)) + half * 8;
            float f0 = 0.f, f1 = 0.f;
            if (gv) { f0 = xr[kk]; f1 = xr[kk + 1]; }
            float h0 = bf_trunc_f(f0), h1 = bf_trunc_f(f1);
            Ah[2 * r]     = __builtin_bit_cast(__bf16, f_to_bfu(h0));
            Ah[2 * r + 1] = __builtin_bit_cast(__bf16, f_to_bfu(h1));
            Al[2 * r]     = __builtin_bit_cast(__bf16, f_to_bfu(f0 - h0));
            Al[2 * r + 1] = __builtin_bit_cast(__bf16, f_to_bfu(f1 - h1));
        }
#pragma unroll
        for (int i = 0; i < 16; ++i) {
            int kk = kb + half * 8 + ((i < 8) ? i : i + 8);
            float w  = Wm[(size_t)kk * Cout + nb + ln];
            float wh = bf_trunc_f(w);
            Bh[i] = __builtin_bit_cast(__bf16, f_to_bfu(wh));
            Bl[i] = __builtin_bit_cast(__bf16, f_to_bfu(w - wh));
        }
        acc = __builtin_amdgcn_wmma_f32_16x16x32_bf16(false, Ah, false, Bh, (short)0, acc, false, false);
        acc = __builtin_amdgcn_wmma_f32_16x16x32_bf16(false, Ah, false, Bl, (short)0, acc, false, false);
        acc = __builtin_amdgcn_wmma_f32_16x16x32_bf16(false, Al, false, Bh, (short)0, acc, false, false);
    }
#pragma unroll
    for (int j = 0; j < 8; ++j) {
        int mrow = m0 + j + 8 * half;
        if (mrow < M)
            out[(size_t)mrow * Cout + nb + ln] = acc[j];
    }
}

// ---------- elementwise kernels ----------
// bn+relu fused with fp32 -> bf16 hi/lo split; emits two ushort planes.
__global__ void bn_relu_bf16_kernel(const float* __restrict__ x, const float* __restrict__ p,
                                    unsigned short* __restrict__ y, size_t plane,
                                    const int* __restrict__ np, int C)
{
    int n = np[0];
    long long i = (long long)blockIdx.x * blockDim.x + threadIdx.x;
    if (i >= (long long)n * C) return;
    int c = (int)(i % C);
    float t = (x[i] - p[2 * C + c]) * (p[c] * rsqrtf(p[3 * C + c] + EPSV)) + p[C + c];
    t = fmaxf(t, 0.f);
    float hi = bf_trunc_f(t);
    y[i]         = f_to_bfu(hi);
    y[plane + i] = f_to_bfu(t - hi);
}

__global__ void zero_kernel(float* __restrict__ y, long long n) {
    long long i = (long long)blockIdx.x * blockDim.x + threadIdx.x;
    if (i < n) y[i] = 0.f;
}

__global__ void copy_kernel(const float* __restrict__ x, float* __restrict__ y, long long n) {
    long long i = (long long)blockIdx.x * blockDim.x + threadIdx.x;
    if (i < n) y[i] = x[i];
}

// y[i, 0:C] = a[i], y[i, C:2C] = b[i]
__global__ void concat_kernel(const float* __restrict__ a, const float* __restrict__ b,
                              float* __restrict__ y, int n, int C) {
    long long i = (long long)blockIdx.x * blockDim.x + threadIdx.x;
    if (i >= (long long)n * 2 * C) return;
    int c = (int)(i % (2 * C));
    long long r = i / (2 * C);
    y[i] = (c < C) ? a[r * C + c] : b[r * C + (c - C)];
}

__global__ void setup_scalars(int* s, int n0, const int* __restrict__ n1p) {
    if (threadIdx.x == 0) { s[0] = n0; s[1] = n1p[0]; }
}

// ---------- host-side helpers ----------
static void run_bn(const float* x, const float* p, void* ybuf, int n0max, int C,
                   const int* np, hipStream_t s) {
    long long tot = (long long)n0max * C;
    bn_relu_bf16_kernel<<<(unsigned)((tot + 255) / 256), 256, 0, s>>>(
        x, p, (unsigned short*)ybuf, (size_t)n0max * C, np, C);
}
static void run_zero(float* y, long long n, hipStream_t s) {
    zero_kernel<<<(unsigned)((n + 255) / 256), 256, 0, s>>>(y, n);
}
static void run_conv(const void* xplanes, int n0max, const unsigned short* wfrag,
                     const int* pi, const int* po, float* out,
                     int P, int K, int Cin, int Cout,
                     const int* nip, const int* nop, hipStream_t s) {
    dim3 g((P + 63) / 64, K, Cout / 16), b(32, 4);
    sparse_conv_wmma<<<g, b, 0, s>>>((const unsigned short*)xplanes, (size_t)n0max * Cin,
                                     wfrag, pi, po, out, P, Cin, Cout, nip, nop);
}
static void run_repack(const float* W, unsigned short* out, int K, int Cin, int Cout, hipStream_t s) {
    long long tot = (long long)K * Cin * Cout;
    repack_w_kernel<<<(unsigned)((tot + 255) / 256), 256, 0, s>>>(W, out, K, Cin, Cout);
}

extern "C" void kernel_launch(void* const* d_in, const int* in_sizes, int n_in_cnt,
                              void* d_out, int out_size, void* d_ws, size_t ws_size,
                              hipStream_t stream)
{
    const float* feats   = (const float*)d_in[0];
    const float* res0_W  = (const float*)d_in[1];
    const float* res0_bn = (const float*)d_in[2];
    const float* down_bn = (const float*)d_in[3];
    const float* down_W  = (const float*)d_in[4];
    const float* res1_W  = (const float*)d_in[5];
    const float* res1_bn = (const float*)d_in[6];
    const float* up_bn   = (const float*)d_in[7];
    const float* up_W    = (const float*)d_in[8];
    const float* t0_bn1  = (const float*)d_in[9];
    const float* t0_W1   = (const float*)d_in[10];
    const float* t0_bn2  = (const float*)d_in[11];
    const float* t0_W2   = (const float*)d_in[12];
    const float* t0_Wsc  = (const float*)d_in[13];
    const float* t1_W    = (const float*)d_in[14];
    const float* t1_bn   = (const float*)d_in[15];
    const int* s0i = (const int*)d_in[16];
    const int* s0o = (const int*)d_in[17];
    const int* s1i = (const int*)d_in[18];
    const int* s1o = (const int*)d_in[19];
    const int* dni = (const int*)d_in[20];
    const int* dno = (const int*)d_in[21];
    const int* n1p = (const int*)d_in[22];

    const int C0 = 32, C1 = 64;
    const int n0 = in_sizes[0] / C0;
    const int P0 = in_sizes[16] / 27;
    const int P1 = in_sizes[18] / 27;
    const int Pd = in_sizes[20] / 8;

    size_t cap = (size_t)n0 * C1;           // n0 x 64 floats per buffer
    float* A  = (float*)d_ws;
    float* Bf = A  + cap;
    float* Cf = Bf + cap;
    float* Df = Cf + cap;

    // weight-fragment area (bf16 hi/lo, fragment-ordered), 64B-aligned slabs
    unsigned short* wf_res0 = (unsigned short*)(Df + cap);
    const size_t sz_res0 = (size_t)108 * 32 * 32 * 2;      // (2,2,27) folded
    unsigned short* wf_down = wf_res0 + sz_res0;
    const size_t sz_down = (size_t)8 * 32 * 64 * 2;
    unsigned short* wf_res1 = wf_down + sz_down;
    const size_t sz_res1 = (size_t)108 * 64 * 64 * 2;
    unsigned short* wf_up   = wf_res1 + sz_res1;
    const size_t sz_up   = (size_t)8 * 64 * 32 * 2;
    unsigned short* wf_t0w1 = wf_up + sz_up;
    const size_t sz_t0w1 = (size_t)27 * 64 * 32 * 2;
    unsigned short* wf_t0w2 = wf_t0w1 + sz_t0w1;
    const size_t sz_t0w2 = (size_t)27 * 32 * 32 * 2;
    unsigned short* wf_t1   = wf_t0w2 + sz_t0w2;
    const size_t sz_t1 = (size_t)54 * 32 * 32 * 2;
    int* scal = (int*)(wf_t1 + sz_t1);

    setup_scalars<<<1, 32, 0, stream>>>(scal, n0, n1p);
    const int* n0p = scal;
    const int* n1d = scal + 1;

    // ---- prologue: repack all conv weights into WMMA fragment order ----
    run_repack(res0_W, wf_res0, 108, C0, C0, stream);
    run_repack(down_W, wf_down,   8, C0, C1, stream);
    run_repack(res1_W, wf_res1, 108, C1, C1, stream);
    run_repack(up_W,   wf_up,     8, C1, C0, stream);
    run_repack(t0_W1,  wf_t0w1,  27, C1, C0, stream);
    run_repack(t0_W2,  wf_t0w2,  27, C0, C0, stream);
    run_repack(t1_W,   wf_t1,    54, C0, C0, stream);

    const size_t kstride00 = (size_t)27 * C0 * C0 * 2;     // per-27-offset slab, ushorts
    const size_t kstride11 = (size_t)27 * C1 * C1 * 2;

    // ---- level 0: x = feats; 2x residual blocks (C0) ----
    copy_kernel<<<(unsigned)(((long long)n0 * C0 + 255) / 256), 256, 0, stream>>>(
        feats, A, (long long)n0 * C0);
    for (int i = 0; i < 2; ++i) {
        const unsigned short* w0 = wf_res0 + (size_t)(i * 2)     * kstride00;
        const unsigned short* w1 = wf_res0 + (size_t)(i * 2 + 1) * kstride00;
        const float* bn = res0_bn + (size_t)i * 2 * 4 * C0;
        run_bn(A, bn, Bf, n0, C0, n0p, stream);
        run_zero(Cf, (long long)n0 * C0, stream);
        run_conv(Bf, n0, w0, s0i, s0o, Cf, P0, 27, C0, C0, n0p, n0p, stream);
        run_bn(Cf, bn + 4 * C0, Bf, n0, C0, n0p, stream);
        run_conv(Bf, n0, w1, s0i, s0o, A, P0, 27, C0, C0, n0p, n0p, stream); // += identity
    }

    // ---- downsample: x1 = conv(bn_relu(x)), C0 -> C1 ----
    run_bn(A, down_bn, Bf, n0, C0, n0p, stream);
    run_zero(Cf, (long long)n0 * C1, stream);
    run_conv(Bf, n0, wf_down, dni, dno, Cf, Pd, 8, C0, C1, n0p, n1d, stream);

    // ---- level 1: 2x residual blocks (C1) on n1 rows ----
    for (int i = 0; i < 2; ++i) {
        const unsigned short* w0 = wf_res1 + (size_t)(i * 2)     * kstride11;
        const unsigned short* w1 = wf_res1 + (size_t)(i * 2 + 1) * kstride11;
        const float* bn = res1_bn + (size_t)i * 2 * 4 * C1;
        run_bn(Cf, bn, Bf, n0, C1, n1d, stream);
        run_zero(Df, (long long)n0 * C1, stream);
        run_conv(Bf, n0, w0, s1i, s1o, Df, P1, 27, C1, C1, n1d, n1d, stream);
        run_bn(Df, bn + 4 * C1, Bf, n0, C1, n1d, stream);
        run_conv(Bf, n0, w1, s1i, s1o, Cf, P1, 27, C1, C1, n1d, n1d, stream);
    }

    // ---- upsample: dec = conv(bn_relu(x1)), C1 -> C0, pairs swapped ----
    run_bn(Cf, up_bn, Bf, n0, C1, n1d, stream);
    run_zero(Df, (long long)n0 * C0, stream);
    run_conv(Bf, n0, wf_up, dno, dni, Df, Pd, 8, C1, C0, n1d, n0p, stream);

    // ---- concat [identity | dec] -> Bf (n0 x 64, f32) ----
    concat_kernel<<<(unsigned)(((long long)n0 * C1 + 255) / 256), 256, 0, stream>>>(
        A, Df, Bf, n0, C0);

    // ---- tail0 ----
    run_bn(Bf, t0_bn1, Cf, n0, C1, n0p, stream);
    run_zero(Df, (long long)n0 * C0, stream);
    run_conv(Cf, n0, wf_t0w1, s0i, s0o, Df, P0, 27, C1, C0, n0p, n0p, stream);
    run_bn(Df, t0_bn2, Cf, n0, C0, n0p, stream);
    {   // A = cat @ Wsc  (dense WMMA GEMM, plain write)
        dim3 g((n0 + 63) / 64, 1, C0 / 16), b(32, 4);
        dense_gemm_wmma<<<g, b, 0, stream>>>(Bf, t0_Wsc, A, n0, C1, C0);
    }
    run_conv(Cf, n0, wf_t0w2, s0i, s0o, A, P0, 27, C0, C0, n0p, n0p, stream);  // += conv2

    // ---- tail1 residual block ----
    run_bn(A, t1_bn, Bf, n0, C0, n0p, stream);
    run_zero(Cf, (long long)n0 * C0, stream);
    run_conv(Bf, n0, wf_t1, s0i, s0o, Cf, P0, 27, C0, C0, n0p, n0p, stream);
    run_bn(Cf, t1_bn + 4 * C0, Bf, n0, C0, n0p, stream);
    run_conv(Bf, n0, wf_t1 + kstride00, s0i, s0o, A, P0, 27, C0, C0, n0p, n0p, stream);

    copy_kernel<<<(unsigned)(((long long)n0 * C0 + 255) / 256), 256, 0, stream>>>(
        A, (float*)d_out, (long long)n0 * C0);
}
